// SparseQKVW8A8OF16DynaLinear_3358664425624
// MI455X (gfx1250) — 8-bit matrix operand order repaired and checked by reading, compiled, not run on hardware
//
#include <hip/hip_runtime.h>
#include <hip/hip_bf16.h>


#define M_DIM 16384
#define K_DIM 4096
#define N_DIM 4096
#define KT    (K_DIM / 64)

typedef __attribute__((ext_vector_type(8))) int v8i;

__device__ __forceinline__ unsigned pack4(const int* __restrict__ p) {
  return ( (unsigned)(p[0] & 0xff)        ) |
         ( (unsigned)(p[1] & 0xff) <<  8  ) |
         ( (unsigned)(p[2] & 0xff) << 16  ) |
         ( (unsigned)(p[3] & 0xff) << 24  );
}

__global__ void __launch_bounds__(256) pack_a_kernel(const int* __restrict__ x,
                                                     char* __restrict__ dst) {
  int tid  = blockIdx.x * blockDim.x + threadIdx.x;
  int lane = tid & 31;
  int f    = tid >> 5;
  if (f >= (M_DIM / 16) * KT) return;
  int kt = f % KT;
  int mt = f / KT;
  int m     = mt * 16 + (lane & 15);
  int kbase = kt * 64 + 8 * (lane >> 4);
  const int* src = x + (size_t)m * K_DIM + kbase;

  unsigned w[8];
#pragma unroll
  for (int g = 0; g < 4; ++g) {
    w[2 * g]     = pack4(src + g * 16);
    w[2 * g + 1] = pack4(src + g * 16 + 4);
  }
  uint4* out = (uint4*)(dst + (size_t)f * 1024 + lane * 32);
  out[0] = make_uint4(w[0], w[1], w[2], w[3]);
  out[1] = make_uint4(w[4], w[5], w[6], w[7]);
}

__global__ void __launch_bounds__(256) pack_b_kernel(const int* __restrict__ wgt,
                                                     char* __restrict__ dst) {
  int tid  = blockIdx.x * blockDim.x + threadIdx.x;
  int lane = tid & 31;
  int f    = tid >> 5;
  if (f >= (N_DIM / 16) * KT) return;
  int kt = f % KT;
  int nt = f / KT;
  int n     = nt * 16 + (lane & 15);
  int kbase = kt * 64 + 8 * (lane >> 4);
  const int* src = wgt + (size_t)n * K_DIM + kbase;

  unsigned q[8];
#pragma unroll
  for (int g = 0; g < 4; ++g) q[2 * g]     = pack4(src + g * 16);
#pragma unroll
  for (int g = 0; g < 4; ++g) q[2 * g + 1] = pack4(src + g * 16 + 4);
  uint4* out = (uint4*)(dst + (size_t)f * 1024 + lane * 32);
  out[0] = make_uint4(q[0], q[1], q[2], q[3]);
  out[1] = make_uint4(q[4], q[5], q[6], q[7]);
}

__global__ void __launch_bounds__(256, 1)
gemm_w8a8_kernel(const char* __restrict__ Ap, const char* __restrict__ Bp,
                 const float* __restrict__ scale_in,
                 const float* __restrict__ scale_w,
                 const int* __restrict__ sum_in,
                 const int* __restrict__ zp_w,
                 const float* __restrict__ bias,
                 float* __restrict__ out) {
  const int lane = threadIdx.x & 31;
  const int wave = threadIdx.x >> 5;
  const int wm   = wave >> 1;
  const int wn   = wave & 1;
  const int m0   = blockIdx.y * 256 + wm * 64;
  const int n0   = blockIdx.x * 128 + wn * 64;

  const char* pA[4];
  const char* pB[4];
#pragma unroll
  for (int i = 0; i < 4; ++i) {
    pA[i] = Ap + ((size_t)(m0 / 16 + i) * KT) * 1024 + lane * 32;
    pB[i] = Bp + ((size_t)(n0 / 16 + i) * KT) * 1024 + lane * 32;
  }

  v8i acc[4][4] = {};

  for (int kt = 0; kt < KT; ++kt) {
    const size_t koff = (size_t)kt * 1024;
    v8i a[4], b[4];
#pragma unroll
    for (int i = 0; i < 4; ++i) a[i] = *(const v8i*)(pA[i] + koff);
#pragma unroll
    for (int i = 0; i < 4; ++i) b[i] = *(const v8i*)(pB[i] + koff);
#pragma unroll
    for (int mi = 0; mi < 4; ++mi)
#pragma unroll
      for (int ni = 0; ni < 4; ++ni)
        acc[mi][ni] = __builtin_amdgcn_wmma_i32_16x16x64_iu8(
             true, a[mi],  true, b[ni],
            acc[mi][ni],  false,  false);
  }

  const int nlo = lane & 15;
  const int lhi = lane >> 4;
#pragma unroll
  for (int ni = 0; ni < 4; ++ni) {
    const int n  = n0 + ni * 16 + nlo;
    const float sw = scale_w[n];
    const float bs = bias[n];
    const int   zp = zp_w[n];
#pragma unroll
    for (int mi = 0; mi < 4; ++mi) {
      const int mbase = m0 + mi * 16 + lhi * 8;
#pragma unroll
      for (int i = 0; i < 8; ++i) {
        const int m = mbase + i;
        const int corrected = acc[mi][ni][i] + sum_in[m] * zp;
        const float f = (float)corrected * scale_in[m] * sw + bs;
        out[(size_t)m * N_DIM + n] = f;
      }
    }
  }
}

extern "C" void kernel_launch(void* const* d_in, const int* in_sizes, int n_in,
                              void* d_out, int out_size, void* d_ws, size_t ws_size,
                              hipStream_t stream) {
  const int*   x        = (const int*)d_in[0];
  const int*   w        = (const int*)d_in[1];
  const float* scale_in = (const float*)d_in[2];
  const float* scale_w  = (const float*)d_in[3];
  const int*   sum_in   = (const int*)d_in[4];
  const int*   zp_w     = (const int*)d_in[5];
  const float* bias     = (const float*)d_in[6];
  float*       out      = (float*)d_out;

  char* wsA = (char*)d_ws;
  char* wsB = wsA + (size_t)M_DIM * K_DIM;

  {
    int threads = (M_DIM / 16) * KT * 32;
    pack_a_kernel<<<threads / 256, 256, 0, stream>>>(x, wsA);
  }
  {
    int threads = (N_DIM / 16) * KT * 32;
    pack_b_kernel<<<threads / 256, 256, 0, stream>>>(w, wsB);
  }
  dim3 grid(N_DIM / 128, M_DIM / 256);
  gemm_w8a8_kernel<<<grid, 256, 0, stream>>>(wsA, wsB, scale_in, scale_w,
                                             sum_in, zp_w, bias, out);
}
